// KM_mdoel_19335942767193
// MI455X (gfx1250) — matrix operand order repaired and checked by reading, not compiled, not run on hardware
//
#include <hip/hip_runtime.h>
#include <cstdint>
#include <cstddef>

#define DI __device__ __forceinline__

typedef __bf16 v16bf __attribute__((ext_vector_type(16)));
typedef float  v8f   __attribute__((ext_vector_type(8)));

union Frag { v16bf v; uint4 u[2]; };

DI unsigned pk2(float a, float b) {
  unsigned short lo = __builtin_bit_cast(unsigned short, (__bf16)a);
  unsigned short hi = __builtin_bit_cast(unsigned short, (__bf16)b);
  return (unsigned)lo | ((unsigned)hi << 16);
}

// Async global -> LDS copy of 16 bytes (CDNA5 path, tracked by ASYNCcnt).
// lds_off is the byte offset from the workgroup LDS base (only dynamic LDS is
// used, so the extern __shared__ array starts at offset 0).
DI void async_g2l_b128(uint32_t lds_off, const void* gsrc) {
  asm volatile("global_load_async_to_lds_b128 %0, %1, off"
               :
               : "v"(lds_off), "v"(gsrc)
               : "memory");
}

DI void wait_async0() {
  asm volatile("s_wait_asynccnt 0x0" ::: "memory");
}

// A-matrix fragment (16x32 bf16): lane holds row (lane&15); lanes 0-15 hold
// K = 0..7 and 16..23, lanes 16-31 hold K = 8..15 and 24..31 (ISA 7.12.2).
DI Frag ldA(const __bf16* base, int stride, int rowbase, int k0, int lane) {
  Frag f;
  const __bf16* p = base + (size_t)(rowbase + (lane & 15)) * stride + k0 +
                    (((lane >> 4) & 1) << 3);
  f.u[0] = *(const uint4*)(p);
  f.u[1] = *(const uint4*)(p + 16);
  return f;
}

// Same fragment, source is f32 (convert to bf16 on the fly).
DI Frag ldA_f32(const float* base, int stride, int rowbase, int k0, int lane) {
  Frag f;
  const float* p = base + (size_t)(rowbase + (lane & 15)) * stride + k0 +
                   (((lane >> 4) & 1) << 3);
  float4 a = *(const float4*)(p);
  float4 b = *(const float4*)(p + 4);
  float4 c = *(const float4*)(p + 16);
  float4 d = *(const float4*)(p + 20);
  f.u[0] = make_uint4(pk2(a.x, a.y), pk2(a.z, a.w), pk2(b.x, b.y), pk2(b.z, b.w));
  f.u[1] = make_uint4(pk2(c.x, c.y), pk2(c.z, c.w), pk2(d.x, d.y), pk2(d.z, d.w));
  return f;
}

// B-matrix fragment (32x16 bf16): lane holds column (lane&15); lanes 0-15 hold
// K = 0..7 and 16..23, lanes 16-31 hold K = 8..15 and 24..31 (matches A).
// Source array is column-major: base[col * stride + k].
DI Frag ldB(const __bf16* base, int stride, int colbase, int k0, int lane) {
  Frag f;
  const __bf16* p = base + (size_t)(colbase + (lane & 15)) * stride + k0 +
                    (((lane >> 4) & 1) << 3);
  f.u[0] = *(const uint4*)(p);
  f.u[1] = *(const uint4*)(p + 16);
  return f;
}

DI v8f wmma_bf16(Frag a, Frag b, v8f c) {
  return __builtin_amdgcn_wmma_f32_16x16x32_bf16(false, a.v, false, b.v,
                                                 (short)0, c, false, false);
}

// ---------------- problem constants ----------------
constexpr int Nent = 32, Dim = 768, Pdim = 512;
constexpr int PLPAD = 3104;  // pooled 3074 zero-padded to 97*32

// ---------------- kernel-1 LDS layout (bytes) ----------------
constexpr int XS = 776, QS = 516, ASd = 516, KSd = 520, VTS = 40, WTS = 40,
              SCS = 33, ATS = 40;
constexpr int OFF_X  = 0;                 // bf16 input [32][776] / f32 align [32][516]
constexpr int OFF_Q  = OFF_X  + 66048;    // f32 q [32][516]
constexpr int OFF_K  = OFF_Q  + 66048;    // bf16 k [32][520]
constexpr int OFF_VT = OFF_K  + 33280;    // bf16 v^T [512][40]
constexpr int OFF_WT = OFF_VT + 40960;    // bf16 weight slice^T [512][40]
constexpr int OFF_SC = OFF_WT + 40960;    // f32 scores [32][33]
constexpr int OFF_AT = OFF_SC + 4224;     // bf16 attn [32][40]
constexpr int OFF_DV = OFF_AT + 2560;     // f32 dot scratch [32]
constexpr int LDS1   = OFF_DV + 128;      // 254208 bytes (<320KB/WGP)

// ---------------- kernel-2 LDS layout ----------------
constexpr int HS = 520;
constexpr int K2_WT = 0;                  // bf16 W1 slice^T [512][40]
constexpr int K2_H  = 40960;              // bf16 h [64][520]
constexpr int K2_W2 = 107520;             // bf16 W2^T [32][520]
constexpr int LDS2  = 140800;

// ---------------- workspace layout (bytes) ----------------
constexpr size_t WS_Q  = 0;
constexpr size_t WS_K  = WS_Q  + (size_t)24 * 512 * 32 * 2;
constexpr size_t WS_V  = WS_K  + (size_t)24 * 512 * 32 * 2;
constexpr size_t WS_W1 = WS_V  + (size_t)24 * 512 * 32 * 2;
constexpr size_t WS_W2 = WS_W1 + (size_t)97 * 512 * 32 * 2;
constexpr size_t WS_PL = WS_W2 + (size_t)512 * 32 * 2;   // pooled bf16 [4096][3104]
// total = WS_PL + 4096*3104*2 = 30,998,528 bytes required in d_ws

// Pack f32 weight [Dk][P] -> bf16 [ceil(Dkpad/32)][P][32] (B-fragment friendly).
__global__ void pack_wT(const float* __restrict__ src, __bf16* __restrict__ dst,
                        int Dk, int Dkpad, int P) {
  int total = Dkpad * P;
  for (int i = blockIdx.x * blockDim.x + threadIdx.x; i < total;
       i += gridDim.x * blockDim.x) {
    int k = i / P, c = i - k * P;
    float f = (k < Dk) ? src[(size_t)k * P + c] : 0.0f;
    dst[((size_t)(k >> 5) * P + c) * 32 + (k & 31)] = (__bf16)f;
  }
}

// Pack W2 f32 [512][32] -> bf16 [32 col][512 k].
__global__ void pack_w2T(const float* __restrict__ src, __bf16* __restrict__ dst) {
  int i = blockIdx.x * blockDim.x + threadIdx.x;
  if (i < 512 * 32) {
    int k = i >> 5, c = i & 31;
    dst[(size_t)c * 512 + k] = (__bf16)src[i];
  }
}

// One projection: out[32][512] = xin[32][768] @ W[768][512] + bias.
// mode 0: f32 row-major out; mode 1: bf16 row-major; mode 2: bf16 transposed.
DI void projection(const __bf16* __restrict__ wsW, const float* __restrict__ bias,
                   const __bf16* xin, __bf16* wT, float* outF, __bf16* outB,
                   int outStride, int mode, int tid, int lane, int wave) {
  const int mtile = wave & 1;
  const int n0 = wave >> 1;
  v8f acc[8];
#pragma unroll
  for (int j = 0; j < 8; ++j)
#pragma unroll
    for (int g = 0; g < 8; ++g) acc[j][g] = 0.0f;

  for (int ks = 0; ks < 24; ++ks) {
    __syncthreads();  // all waves done reading previous wT slice
    // async-stage pre-packed bf16 weight slice (32KB) directly into LDS
    const __bf16* src = wsW + (size_t)ks * (512 * 32);
#pragma unroll
    for (int i = 0; i < 8; ++i) {
      int u = tid + i * 256;  // 2048 x 16B
      int c = u >> 2, k8 = (u & 3) << 3;
      async_g2l_b128((uint32_t)(OFF_WT + (c * WTS + k8) * 2), src + u * 8);
    }
    wait_async0();
    __syncthreads();
    Frag a = ldA(xin, XS, mtile * 16, ks * 32, lane);
#pragma unroll
    for (int j = 0; j < 8; ++j) {
      Frag bf = ldB(wT, WTS, (n0 + 4 * j) * 16, 0, lane);
      acc[j] = wmma_bf16(a, bf, acc[j]);
    }
  }
  const int r0 = ((lane >> 4) & 1) << 3;
#pragma unroll
  for (int j = 0; j < 8; ++j) {
    int c = (n0 + 4 * j) * 16 + (lane & 15);
    float bv = bias[c];
#pragma unroll
    for (int g = 0; g < 8; ++g) {
      float val = acc[j][g] + bv;
      int r = mtile * 16 + r0 + g;
      if (mode == 0)      outF[(size_t)r * outStride + c] = val;
      else if (mode == 1) outB[(size_t)r * outStride + c] = (__bf16)val;
      else                outB[(size_t)c * outStride + r] = (__bf16)val;
    }
  }
}

// Kernel 1: one workgroup per batch item. Fully fused attention + fusion + pool.
__global__ __launch_bounds__(256, 1)
void fused_attn_kernel(const float* __restrict__ content,
                       const float* __restrict__ image,
                       const __bf16* __restrict__ wsQ, const __bf16* __restrict__ wsK,
                       const __bf16* __restrict__ wsV,
                       const float* __restrict__ bq, const float* __restrict__ bk,
                       const float* __restrict__ bv,
                       __bf16* __restrict__ pooled) {
  extern __shared__ char smem[];
  __bf16* xin = (__bf16*)(smem + OFF_X);
  float*  alg = (float*)(smem + OFF_X);   // reuses input region after k/v done
  float*  q   = (float*)(smem + OFF_Q);
  __bf16* kl  = (__bf16*)(smem + OFF_K);
  __bf16* vT  = (__bf16*)(smem + OFF_VT);
  __bf16* wT  = (__bf16*)(smem + OFF_WT);
  float*  sc  = (float*)(smem + OFF_SC);
  __bf16* at  = (__bf16*)(smem + OFF_AT);
  float*  dv  = (float*)(smem + OFF_DV);

  const int tid = threadIdx.x, lane = tid & 31, wave = tid >> 5;
  const int b = blockIdx.x;

  {  // stage content f32 -> bf16 LDS
    const float* cb = content + (size_t)b * (Nent * Dim);
    for (int i = tid; i < (Nent * Dim) / 4; i += 256) {
      int e = i * 4, r = e / Dim, c = e % Dim;
      float4 x = *(const float4*)(cb + e);
      *(uint2*)(xin + r * XS + c) = make_uint2(pk2(x.x, x.y), pk2(x.z, x.w));
    }
  }
  projection(wsQ, bq, xin, wT, q, nullptr, QS, 0, tid, lane, wave);
  __syncthreads();
  {  // stage image f32 -> bf16 LDS (overwrite)
    const float* ib = image + (size_t)b * (Nent * Dim);
    for (int i = tid; i < (Nent * Dim) / 4; i += 256) {
      int e = i * 4, r = e / Dim, c = e % Dim;
      float4 x = *(const float4*)(ib + e);
      *(uint2*)(xin + r * XS + c) = make_uint2(pk2(x.x, x.y), pk2(x.z, x.w));
    }
  }
  projection(wsK, bk, xin, wT, nullptr, kl, KSd, 1, tid, lane, wave);
  projection(wsV, bv, xin, wT, nullptr, vT, VTS, 2, tid, lane, wave);
  __syncthreads();

  // scores = q @ k^T  (K = 512, 16 ksteps); 4 output tiles -> waves 0..3
  if (wave < 4) {
    int mt = wave & 1, nt = wave >> 1;
    v8f s;
#pragma unroll
    for (int g = 0; g < 8; ++g) s[g] = 0.0f;
    for (int ks = 0; ks < 16; ++ks) {
      Frag a  = ldA_f32(q, QS, mt * 16, ks * 32, lane);
      Frag bf = ldB(kl, KSd, nt * 16, ks * 32, lane);
      s = wmma_bf16(a, bf, s);
    }
    int r0 = ((lane >> 4) & 1) << 3, c = nt * 16 + (lane & 15);
#pragma unroll
    for (int g = 0; g < 8; ++g) sc[(mt * 16 + r0 + g) * SCS + c] = s[g];
  }
  __syncthreads();

  // softmax over rows (scale 1/sqrt(512)), write attn as bf16
  if (tid < 32) {
    const float scale = 0.044194173824159216f;
    float mx = -3.402823466e38f;
    for (int m = 0; m < 32; ++m) mx = fmaxf(mx, sc[tid * SCS + m] * scale);
    float ssum = 0.0f;
    for (int m = 0; m < 32; ++m) {
      float e = __expf(sc[tid * SCS + m] * scale - mx);
      ssum += e;
      sc[tid * SCS + m] = e;
    }
    float inv = 1.0f / ssum;
    for (int m = 0; m < 32; ++m) at[tid * ATS + m] = (__bf16)(sc[tid * SCS + m] * inv);
  }
  __syncthreads();

  // align = attn @ v  (single K=32 step; vT gives contiguous B fragments)
  {
    int mt = wave & 1, n0 = wave >> 1;
    Frag a = ldA(at, ATS, mt * 16, 0, lane);
    int r0 = ((lane >> 4) & 1) << 3;
#pragma unroll
    for (int j = 0; j < 8; ++j) {
      int nt = n0 + 4 * j;
      Frag bf = ldB(vT, VTS, nt * 16, 0, lane);
      v8f s;
#pragma unroll
      for (int g = 0; g < 8; ++g) s[g] = 0.0f;
      s = wmma_bf16(a, bf, s);
      int c = nt * 16 + (lane & 15);
#pragma unroll
      for (int g = 0; g < 8; ++g) alg[(size_t)(mt * 16 + r0 + g) * ASd + c] = s[g];
    }
  }
  __syncthreads();

  // fusion [q, align, q-align, dot] -> avg & max pool over the 32 entities
  __bf16* prow = pooled + (size_t)b * PLPAD;
  for (int f = tid; f < 1536; f += 256) {
    int seg = f >> 9, p = f & 511;
    float sum = 0.0f, mx = -3.402823466e38f;
#pragma unroll 4
    for (int n = 0; n < 32; ++n) {
      float qv = q[n * QS + p], av = alg[n * ASd + p];
      float val = (seg == 0) ? qv : (seg == 1) ? av : (qv - av);
      sum += val;
      mx = fmaxf(mx, val);
    }
    prow[f]        = (__bf16)(sum * (1.0f / 32.0f));
    prow[1537 + f] = (__bf16)mx;
  }
  if (tid < 32) {  // dot feature (index 1536)
    float d = 0.0f;
    for (int p = 0; p < 512; ++p) d += q[tid * QS + p] * alg[tid * ASd + p];
    dv[tid] = d;
  }
  __syncthreads();
  if (tid == 0) {
    float s = 0.0f, mx = -3.402823466e38f;
    for (int n = 0; n < 32; ++n) { s += dv[n]; mx = fmaxf(mx, dv[n]); }
    prow[1536]        = (__bf16)(s * (1.0f / 32.0f));
    prow[1537 + 1536] = (__bf16)mx;
  }
  if (tid < PLPAD - 3074) prow[3074 + tid] = (__bf16)0.0f;  // zero K pad
}

// Kernel 2: FFN GEMM [4096,3104]@[3104,512] -> relu -> @[512,32] -> BN.
__global__ __launch_bounds__(512, 1)
void ffn_kernel(const __bf16* __restrict__ pooled, const __bf16* __restrict__ wsW1,
                const __bf16* __restrict__ wsW2,
                const float* __restrict__ b1, const float* __restrict__ b2,
                const float* __restrict__ gma, const float* __restrict__ bta,
                const float* __restrict__ mu, const float* __restrict__ var,
                float* __restrict__ out) {
  extern __shared__ char smem[];
  __bf16* wT  = (__bf16*)(smem + K2_WT);
  __bf16* h   = (__bf16*)(smem + K2_H);
  __bf16* w2T = (__bf16*)(smem + K2_W2);
  const int tid = threadIdx.x, lane = tid & 31, wave = tid >> 5;
  const int rb = blockIdx.x * 64;

  // async-stage W2^T into LDS (stride-padded); waited inside first ks iter
  for (int i = tid; i < (32 * 512) / 8; i += 512) {
    int c = i >> 6, k8 = (i & 63) << 3;
    async_g2l_b128((uint32_t)(K2_W2 + (c * HS + k8) * 2), wsW2 + c * 512 + k8);
  }

  const int mt = wave & 3;    // 4 row tiles (64 batch rows)
  const int n0 = wave >> 2;   // 32 col tiles / 4 wave groups
  v8f acc[8];
#pragma unroll
  for (int j = 0; j < 8; ++j)
#pragma unroll
    for (int g = 0; g < 8; ++g) acc[j][g] = 0.0f;

  for (int ks = 0; ks < 97; ++ks) {
    __syncthreads();  // all waves done reading previous wT slice
    const __bf16* src = wsW1 + (size_t)ks * (512 * 32);
#pragma unroll
    for (int i = 0; i < 4; ++i) {
      int u = tid + i * 512;
      int c = u >> 2, k8 = (u & 3) << 3;
      async_g2l_b128((uint32_t)(K2_WT + (c * WTS + k8) * 2), src + u * 8);
    }
    wait_async0();
    __syncthreads();
    Frag a;  // A fragment straight from global bf16 pooled matrix
    {
      const __bf16* p = pooled + (size_t)(rb + mt * 16 + (lane & 15)) * PLPAD +
                        ks * 32 + (((lane >> 4) & 1) << 3);
      a.u[0] = *(const uint4*)(p);
      a.u[1] = *(const uint4*)(p + 16);
    }
#pragma unroll
    for (int j = 0; j < 8; ++j) {
      Frag bf = ldB(wT, WTS, (n0 + 4 * j) * 16, 0, lane);
      acc[j] = wmma_bf16(a, bf, acc[j]);
    }
  }

  // bias + relu -> h (bf16, row-major)
  const int r0 = ((lane >> 4) & 1) << 3;
#pragma unroll
  for (int j = 0; j < 8; ++j) {
    int c = (n0 + 4 * j) * 16 + (lane & 15);
    float bv = b1[c];
#pragma unroll
    for (int g = 0; g < 8; ++g) {
      float vH = fmaxf(acc[j][g] + bv, 0.0f);
      h[(size_t)(mt * 16 + r0 + g) * HS + c] = (__bf16)vH;
    }
  }
  __syncthreads();

  // y = h @ W2 + b2, then eval-mode BN; 8 output tiles -> waves 0..7
  if (wave < 8) {
    int mt2 = wave & 3, nt2 = wave >> 2;
    v8f s;
#pragma unroll
    for (int g = 0; g < 8; ++g) s[g] = 0.0f;
    for (int ks = 0; ks < 16; ++ks) {
      Frag a  = ldA(h, HS, mt2 * 16, ks * 32, lane);
      Frag bf = ldB(w2T, HS, nt2 * 16, ks * 32, lane);
      s = wmma_bf16(a, bf, s);
    }
    int c = nt2 * 16 + (lane & 15);
    float scl = gma[c] * rsqrtf(var[c] + 1e-5f);
    float ofs = bta[c] - mu[c] * scl;
    float bb = b2[c];
#pragma unroll
    for (int g = 0; g < 8; ++g) {
      float y = (s[g] + bb) * scl + ofs;
      out[(size_t)(rb + mt2 * 16 + r0 + g) * 32 + c] = y;
    }
  }
}

extern "C" void kernel_launch(void* const* d_in, const int* in_sizes, int n_in,
                              void* d_out, int out_size, void* d_ws, size_t ws_size,
                              hipStream_t stream) {
  (void)in_sizes; (void)n_in; (void)out_size; (void)ws_size;
  const float* content = (const float*)d_in[0];
  const float* image   = (const float*)d_in[1];
  const float* Wq = (const float*)d_in[2];
  const float* bq = (const float*)d_in[3];
  const float* Wk = (const float*)d_in[4];
  const float* bk = (const float*)d_in[5];
  const float* Wv = (const float*)d_in[6];
  const float* bv = (const float*)d_in[7];
  const float* W1 = (const float*)d_in[8];
  const float* b1 = (const float*)d_in[9];
  const float* W2 = (const float*)d_in[10];
  const float* b2 = (const float*)d_in[11];
  const float* gma = (const float*)d_in[12];
  const float* bta = (const float*)d_in[13];
  const float* mu  = (const float*)d_in[14];
  const float* var = (const float*)d_in[15];
  float* out = (float*)d_out;
  char* ws = (char*)d_ws;

  __bf16* wsQ  = (__bf16*)(ws + WS_Q);
  __bf16* wsK  = (__bf16*)(ws + WS_K);
  __bf16* wsV  = (__bf16*)(ws + WS_V);
  __bf16* wsW1 = (__bf16*)(ws + WS_W1);
  __bf16* wsW2 = (__bf16*)(ws + WS_W2);
  __bf16* wsPL = (__bf16*)(ws + WS_PL);

  hipFuncSetAttribute(reinterpret_cast<const void*>(fused_attn_kernel),
                      hipFuncAttributeMaxDynamicSharedMemorySize, LDS1);
  hipFuncSetAttribute(reinterpret_cast<const void*>(ffn_kernel),
                      hipFuncAttributeMaxDynamicSharedMemorySize, LDS2);

  pack_wT<<<512, 256, 0, stream>>>(Wq, wsQ, 768, 768, 512);
  pack_wT<<<512, 256, 0, stream>>>(Wk, wsK, 768, 768, 512);
  pack_wT<<<512, 256, 0, stream>>>(Wv, wsV, 768, 768, 512);
  pack_wT<<<1024, 256, 0, stream>>>(W1, wsW1, 3074, 3104, 512);
  pack_w2T<<<64, 256, 0, stream>>>(W2, wsW2);

  fused_attn_kernel<<<4096, 256, LDS1, stream>>>(content, image, wsQ, wsK, wsV,
                                                 bq, bk, bv, wsPL);
  ffn_kernel<<<64, 512, LDS2, stream>>>(wsPL, wsW1, wsW2, b1, b2, gma, bta, mu,
                                        var, out);
}
